// MoE_16011638079992
// MI455X (gfx1250) — matrix operand order repaired and checked by reading, not compiled, not run on hardware
//
#include <hip/hip_runtime.h>
#include <hip/hip_bf16.h>
#include <math.h>

// Problem constants (match reference)
#define T_TOK 8192   // B*S
#define H_DIM 1024
#define I_DIM 4096
#define N_EXP 8

typedef __attribute__((ext_vector_type(16))) __bf16 v16bf;
typedef __attribute__((ext_vector_type(8)))  __bf16 v8bf;
typedef __attribute__((ext_vector_type(8)))  float  v8f;
typedef __attribute__((ext_vector_type(4)))  float  v4f;

// ---- WMMA fragment loaders (bf16 16x16x32, wave32, per ISA 7.12.2) -------
// A (16x32, MxK): lanes 0-15 -> row=lane, K {0..7} and {16..23};
//                 lanes 16-31 -> row=lane-16, K {8..15} and {24..31}.
__device__ __forceinline__ v16bf frag_a(const __bf16* base, int ldr) {
  const int lane = threadIdx.x & 31;
  const __bf16* p = base + (lane & 15) * ldr + ((lane < 16) ? 0 : 8);
  v8bf lo = *reinterpret_cast<const v8bf*>(p);        // 16B aligned
  v8bf hi = *reinterpret_cast<const v8bf*>(p + 16);   // +32B
  return __builtin_shufflevector(lo, hi, 0,1,2,3,4,5,6,7,8,9,10,11,12,13,14,15);
}
// B (32x16, KxN): lanes 0-15 -> col=lane, K {0..7} and {16..23}; lanes 16-31 -> K {8..15} and {24..31}.
__device__ __forceinline__ v16bf frag_b(const __bf16* base, int ldr) {
  const int lane = threadIdx.x & 31;
  const __bf16* p = base + (lane & 15) * ldr + ((lane < 16) ? 0 : 8);
  v8bf lo = *reinterpret_cast<const v8bf*>(p);
  v8bf hi = *reinterpret_cast<const v8bf*>(p + 16);
  return __builtin_shufflevector(lo, hi, 0,1,2,3,4,5,6,7,8,9,10,11,12,13,14,15);
}

__device__ __forceinline__ v8f v8f_zero() {
  v8f z;
#pragma unroll
  for (int i = 0; i < 8; ++i) z[i] = 0.0f;
  return z;
}

__device__ __forceinline__ void cvt16(const v4f& f0, const v4f& f1,
                                      const v4f& f2, const v4f& f3,
                                      v8bf& b0, v8bf& b1) {
#pragma unroll
  for (int i = 0; i < 4; ++i) {
    b0[i] = (__bf16)f0[i]; b0[4 + i] = (__bf16)f1[i];
    b1[i] = (__bf16)f2[i]; b1[4 + i] = (__bf16)f3[i];
  }
}

// ---- utility kernels ------------------------------------------------------
__global__ void zerof(float* __restrict__ p, int n) {
  int i = blockIdx.x * blockDim.x + threadIdx.x;
  if (i < n) p[i] = 0.0f;
}

__global__ void cvt_bf16(const float* __restrict__ x, __bf16* __restrict__ y, int n4) {
  int i = blockIdx.x * blockDim.x + threadIdx.x;
  if (i < n4) {
    v4f v = *reinterpret_cast<const v4f*>(x + (size_t)i * 4);
#pragma unroll
    for (int j = 0; j < 4; ++j) y[(size_t)i * 4 + j] = (__bf16)v[j];
  }
}

// ---- router + gather: one wave32 per token -------------------------------
// Builds per-expert compacted token lists (index + combine weight) so the
// expert GEMMs only touch routed tokens (~T*K/E rows instead of T).
__global__ __launch_bounds__(256) void moe_router(
    const float* __restrict__ X, const float* __restrict__ GW,
    int* __restrict__ Eidx, float* __restrict__ Ew,
    int* __restrict__ Ecnt, float* __restrict__ Pacc) {
  const int lane = threadIdx.x & 31;
  const int wave = threadIdx.x >> 5;
  const int t = blockIdx.x * 8 + wave;
  const float* xr = X + (size_t)t * H_DIM;

  float acc[N_EXP];
#pragma unroll
  for (int e = 0; e < N_EXP; ++e) acc[e] = 0.0f;

  for (int k = lane; k < H_DIM; k += 32) {
    float xv = xr[k];
#pragma unroll
    for (int e = 0; e < N_EXP; ++e) acc[e] += xv * GW[e * H_DIM + k];
  }
#pragma unroll
  for (int e = 0; e < N_EXP; ++e)
    for (int off = 16; off > 0; off >>= 1) acc[e] += __shfl_xor(acc[e], off, 32);

  if (lane == 0) {
    float m = acc[0];
#pragma unroll
    for (int e = 1; e < N_EXP; ++e) m = fmaxf(m, acc[e]);
    float p[N_EXP], s = 0.0f;
#pragma unroll
    for (int e = 0; e < N_EXP; ++e) { p[e] = __expf(acc[e] - m); s += p[e]; }
    float inv = 1.0f / s;
#pragma unroll
    for (int e = 0; e < N_EXP; ++e) p[e] *= inv;

    int i0 = 0;
#pragma unroll
    for (int e = 1; e < N_EXP; ++e) if (p[e] > p[i0]) i0 = e;
    int i1 = (i0 == 0) ? 1 : 0;
#pragma unroll
    for (int e = 0; e < N_EXP; ++e) if (e != i0 && p[e] > p[i1]) i1 = e;

    float wn = 1.0f / (p[i0] + p[i1]);
    float w0 = p[i0] * wn, w1 = p[i1] * wn;

    int p0 = atomicAdd(&Ecnt[i0], 1);
    Eidx[(size_t)i0 * T_TOK + p0] = t;
    Ew[(size_t)i0 * T_TOK + p0] = w0;
    int p1 = atomicAdd(&Ecnt[i1], 1);
    Eidx[(size_t)i1 * T_TOK + p1] = t;
    Ew[(size_t)i1 * T_TOK + p1] = w1;

#pragma unroll
    for (int e = 0; e < N_EXP; ++e) atomicAdd(&Pacc[e], p[e]);
  }
}

__global__ void moe_aux(const int* __restrict__ Ecnt, const float* __restrict__ Pacc,
                        float* __restrict__ out) {
  if (threadIdx.x == 0 && blockIdx.x == 0) {
    const float invT = 1.0f / (float)T_TOK;
    float s = 0.0f;
#pragma unroll
    for (int e = 0; e < N_EXP; ++e) s += ((float)Ecnt[e] * invT) * (Pacc[e] * invT);
    *out = (float)N_EXP * s;
  }
}

// ---- GEMM1 (gathered): hid[slot] = gelu(x[idx[slot]] @ w1[e]^T) ----------
// M = n_e (compacted), N = I, K = H. Double-buffered LDS, 1 barrier / K-step.
__global__ __launch_bounds__(256) void moe_gemm1(
    const __bf16* __restrict__ Xb, const float* __restrict__ W1,
    const int* __restrict__ Eidx, const int* __restrict__ Ecnt,
    int expert, __bf16* __restrict__ Hid) {
  constexpr int LDR = 40;  // padded bf16 row (conflict-free b128 frag loads)
  __shared__ __bf16 sA[2][128 * LDR];
  __shared__ __bf16 sB[2][128 * LDR];
  __shared__ int sIdx[128];

  const int ne = Ecnt[expert];
  const int m0 = blockIdx.y * 128;
  if (m0 >= ne) return;  // static worst-case grid; only ~n_e/128 blocks work
  const int n0 = blockIdx.x * 128;
  const int tid = threadIdx.x;

  if (tid < 128) {
    int slot = m0 + tid;
    sIdx[tid] = (slot < ne) ? Eidx[(size_t)expert * T_TOK + slot] : 0;
  }
  __syncthreads();

  const int lane = tid & 31;
  const int wave = tid >> 5;
  const int wm = (wave & 1) * 64;
  const int wn = (wave >> 1) * 32;
  const int ldrow = tid >> 1;
  const int ldcol = (tid & 1) * 16;

  const __bf16* Arow = Xb + (size_t)sIdx[ldrow] * H_DIM + ldcol;  // gathered
  const float* Brow = W1 + (size_t)(n0 + ldrow) * H_DIM + ldcol;

  v8f acc[4][2];
#pragma unroll
  for (int mi = 0; mi < 4; ++mi) { acc[mi][0] = v8f_zero(); acc[mi][1] = v8f_zero(); }

  // prologue: stage k=0 into buffer 0
  {
    const v8bf* sa = reinterpret_cast<const v8bf*>(Arow);
    v8bf a0 = sa[0], a1 = sa[1];
    const v4f* sb = reinterpret_cast<const v4f*>(Brow);
    v4f f0 = sb[0], f1 = sb[1], f2 = sb[2], f3 = sb[3];
    v8bf b0, b1;
    cvt16(f0, f1, f2, f3, b0, b1);
    v8bf* dA = reinterpret_cast<v8bf*>(&sA[0][ldrow * LDR + ldcol]);
    dA[0] = a0; dA[1] = a1;
    v8bf* dB = reinterpret_cast<v8bf*>(&sB[0][ldrow * LDR + ldcol]);
    dB[0] = b0; dB[1] = b1;
  }
  __syncthreads();

  constexpr int NSTEP = H_DIM / 32;
  for (int s = 0; s < NSTEP; ++s) {
    const int cur = s & 1;
    const int nxt = cur ^ 1;
    const bool hasNext = (s + 1 < NSTEP);

    v8bf na0, na1, nb0, nb1;
    if (hasNext) {  // issue next tile's global loads before compute
      const int k = (s + 1) * 32;
      const v8bf* sa = reinterpret_cast<const v8bf*>(Arow + k);
      na0 = sa[0]; na1 = sa[1];
      const v4f* sb = reinterpret_cast<const v4f*>(Brow + k);
      v4f f0 = sb[0], f1 = sb[1], f2 = sb[2], f3 = sb[3];
      cvt16(f0, f1, f2, f3, nb0, nb1);
      if (s + 2 < NSTEP) {
        __builtin_prefetch(Arow + (s + 2) * 32, 0, 1);
        __builtin_prefetch(Brow + (s + 2) * 32, 0, 1);
      }
    }

    v16bf af[4], bfb[2];
#pragma unroll
    for (int mi = 0; mi < 4; ++mi) af[mi] = frag_a(&sA[cur][(wm + mi * 16) * LDR], LDR);
#pragma unroll
    for (int ni = 0; ni < 2; ++ni) bfb[ni] = frag_b(&sB[cur][(wn + ni * 16) * LDR], LDR);
#pragma unroll
    for (int mi = 0; mi < 4; ++mi)
#pragma unroll
      for (int ni = 0; ni < 2; ++ni)
        acc[mi][ni] = __builtin_amdgcn_wmma_f32_16x16x32_bf16(
            false, af[mi], false, bfb[ni], (short)0, acc[mi][ni], false, false);

    if (hasNext) {  // fill the other buffer while this one is consumed
      v8bf* dA = reinterpret_cast<v8bf*>(&sA[nxt][ldrow * LDR + ldcol]);
      dA[0] = na0; dA[1] = na1;
      v8bf* dB = reinterpret_cast<v8bf*>(&sB[nxt][ldrow * LDR + ldcol]);
      dB[0] = nb0; dB[1] = nb1;
    }
    __syncthreads();
  }

  const int rb = (lane < 16) ? 0 : 8;
  const int col = lane & 15;
#pragma unroll
  for (int mi = 0; mi < 4; ++mi)
#pragma unroll
    for (int ni = 0; ni < 2; ++ni)
#pragma unroll
      for (int r = 0; r < 8; ++r) {
        int slot = m0 + wm + mi * 16 + rb + r;  // compacted row
        int c = n0 + wn + ni * 16 + col;
        float v = acc[mi][ni][r];
        float g = 0.5f * v * (1.0f + erff(v * 0.70710678118654752f));  // exact gelu
        Hid[(size_t)slot * I_DIM + c] = (__bf16)g;
      }
}

// ---- GEMM2 (scatter): out[idx[slot]] += w[slot] * (hid[slot] @ w2[e]^T) --
__global__ __launch_bounds__(256) void moe_gemm2(
    const __bf16* __restrict__ Hid, const float* __restrict__ W2,
    const int* __restrict__ Eidx, const float* __restrict__ Ew,
    const int* __restrict__ Ecnt, int expert, float* __restrict__ Out) {
  constexpr int LDR = 40;
  __shared__ __bf16 sA[2][128 * LDR];
  __shared__ __bf16 sB[2][128 * LDR];
  __shared__ int sTok[128];
  __shared__ float sW[128];

  const int ne = Ecnt[expert];
  const int m0 = blockIdx.y * 128;
  if (m0 >= ne) return;
  const int n0 = blockIdx.x * 128;
  const int tid = threadIdx.x;

  if (tid < 128) {
    int slot = m0 + tid;
    bool valid = slot < ne;
    sTok[tid] = valid ? Eidx[(size_t)expert * T_TOK + slot] : 0;
    sW[tid] = valid ? Ew[(size_t)expert * T_TOK + slot] : 0.0f;
  }
  __syncthreads();

  const int lane = tid & 31;
  const int wave = tid >> 5;
  const int wm = (wave & 1) * 64;
  const int wn = (wave >> 1) * 32;
  const int ldrow = tid >> 1;
  const int ldcol = (tid & 1) * 16;

  const __bf16* Arow = Hid + (size_t)(m0 + ldrow) * I_DIM + ldcol;  // compact
  const float* Brow = W2 + (size_t)(n0 + ldrow) * I_DIM + ldcol;

  v8f acc[4][2];
#pragma unroll
  for (int mi = 0; mi < 4; ++mi) { acc[mi][0] = v8f_zero(); acc[mi][1] = v8f_zero(); }

  {
    const v8bf* sa = reinterpret_cast<const v8bf*>(Arow);
    v8bf a0 = sa[0], a1 = sa[1];
    const v4f* sb = reinterpret_cast<const v4f*>(Brow);
    v4f f0 = sb[0], f1 = sb[1], f2 = sb[2], f3 = sb[3];
    v8bf b0, b1;
    cvt16(f0, f1, f2, f3, b0, b1);
    v8bf* dA = reinterpret_cast<v8bf*>(&sA[0][ldrow * LDR + ldcol]);
    dA[0] = a0; dA[1] = a1;
    v8bf* dB = reinterpret_cast<v8bf*>(&sB[0][ldrow * LDR + ldcol]);
    dB[0] = b0; dB[1] = b1;
  }
  __syncthreads();

  constexpr int NSTEP = I_DIM / 32;
  for (int s = 0; s < NSTEP; ++s) {
    const int cur = s & 1;
    const int nxt = cur ^ 1;
    const bool hasNext = (s + 1 < NSTEP);

    v8bf na0, na1, nb0, nb1;
    if (hasNext) {
      const int k = (s + 1) * 32;
      const v8bf* sa = reinterpret_cast<const v8bf*>(Arow + k);
      na0 = sa[0]; na1 = sa[1];
      const v4f* sb = reinterpret_cast<const v4f*>(Brow + k);
      v4f f0 = sb[0], f1 = sb[1], f2 = sb[2], f3 = sb[3];
      cvt16(f0, f1, f2, f3, nb0, nb1);
      if (s + 2 < NSTEP) {
        __builtin_prefetch(Arow + (s + 2) * 32, 0, 1);
        __builtin_prefetch(Brow + (s + 2) * 32, 0, 1);
      }
    }

    v16bf af[4], bfb[2];
#pragma unroll
    for (int mi = 0; mi < 4; ++mi) af[mi] = frag_a(&sA[cur][(wm + mi * 16) * LDR], LDR);
#pragma unroll
    for (int ni = 0; ni < 2; ++ni) bfb[ni] = frag_b(&sB[cur][(wn + ni * 16) * LDR], LDR);
#pragma unroll
    for (int mi = 0; mi < 4; ++mi)
#pragma unroll
      for (int ni = 0; ni < 2; ++ni)
        acc[mi][ni] = __builtin_amdgcn_wmma_f32_16x16x32_bf16(
            false, af[mi], false, bfb[ni], (short)0, acc[mi][ni], false, false);

    if (hasNext) {
      v8bf* dA = reinterpret_cast<v8bf*>(&sA[nxt][ldrow * LDR + ldcol]);
      dA[0] = na0; dA[1] = na1;
      v8bf* dB = reinterpret_cast<v8bf*>(&sB[nxt][ldrow * LDR + ldcol]);
      dB[0] = nb0; dB[1] = nb1;
    }
    __syncthreads();
  }

  const int rb = (lane < 16) ? 0 : 8;
  const int col = lane & 15;
#pragma unroll
  for (int mi = 0; mi < 4; ++mi)
#pragma unroll
    for (int ni = 0; ni < 2; ++ni)
#pragma unroll
      for (int r = 0; r < 8; ++r) {
        int lrow = wm + mi * 16 + rb + r;
        if (m0 + lrow < ne) {  // skip tail padding rows
          int tok = sTok[lrow];
          int c = n0 + wn + ni * 16 + col;
          // tokens unique within one expert list + experts sequential on the
          // stream -> plain read-modify-write accumulate is race-free
          Out[(size_t)tok * H_DIM + c] += sW[lrow] * acc[mi][ni][r];
        }
      }
}

// ---- host-side launcher ---------------------------------------------------
extern "C" void kernel_launch(void* const* d_in, const int* in_sizes, int n_in,
                              void* d_out, int out_size, void* d_ws, size_t ws_size,
                              hipStream_t stream) {
  (void)in_sizes; (void)n_in; (void)out_size; (void)ws_size;
  const float* x  = (const float*)d_in[0];   // [T, H]
  const float* gw = (const float*)d_in[1];   // [E, H]
  const float* w1 = (const float*)d_in[2];   // [E, I, H]
  const float* w2 = (const float*)d_in[3];   // [E, H, I]
  float* out = (float*)d_out;                // [T*H] + 1 (aux loss)

  // workspace layout
  char* ws = (char*)d_ws;
  const size_t XB = (size_t)T_TOK * H_DIM * sizeof(__bf16);  // 16 MB
  const size_t HB = (size_t)T_TOK * I_DIM * sizeof(__bf16);  // 64 MB
  const size_t LB = (size_t)T_TOK * N_EXP * sizeof(int);     // 256 KB
  __bf16* xb   = (__bf16*)ws;
  __bf16* hid  = (__bf16*)(ws + XB);
  int*    eidx = (int*)(ws + XB + HB);          // [E, T] gathered token ids
  float*  ew   = (float*)(ws + XB + HB + LB);   // [E, T] gathered weights
  int*    ecnt = (int*)(ws + XB + HB + 2 * LB); // [E]
  float*  pacc = (float*)(ecnt + N_EXP);        // [E], contiguous after ecnt

  const int nOut = T_TOK * H_DIM;
  zerof<<<(nOut + 255) / 256, 256, 0, stream>>>(out, nOut);
  zerof<<<1, 32, 0, stream>>>((float*)ecnt, 2 * N_EXP);  // ecnt + pacc

  cvt_bf16<<<(nOut / 4 + 255) / 256, 256, 0, stream>>>(x, xb, nOut / 4);

  moe_router<<<T_TOK / 8, 256, 0, stream>>>(x, gw, eidx, ew, ecnt, pacc);
  moe_aux<<<1, 32, 0, stream>>>(ecnt, pacc, out + (size_t)nOut);

  for (int e = 0; e < N_EXP; ++e) {
    moe_gemm1<<<dim3(I_DIM / 128, T_TOK / 128), 256, 0, stream>>>(
        xb, w1 + (size_t)e * I_DIM * H_DIM, eidx, ecnt, e, hid);
    moe_gemm2<<<dim3(H_DIM / 128, T_TOK / 128), 256, 0, stream>>>(
        hid, w2 + (size_t)e * H_DIM * I_DIM, eidx, ew, ecnt, e, out);
  }
}
